// Local_2D_Map_Constructor_78022375899663
// MI455X (gfx1250) — compile-verified, matrix operand order checked by reading, not run on hardware
//
#include <hip/hip_runtime.h>
#include <cmath>

// ---------------- constants ----------------
#define B_      8
#define C_      20        // 4 + 16 sem
#define F_      17        // 1 + 16 splatted feature channels
#define FH      480
#define FW      640
#define M_      240
#define VR_     100
#define ZH      80        // MAX_H - MIN_H
#define VOX_PER_CH 800000L          // 100*100*80
#define VOX_FLOATS (8L*17L*VOX_PER_CH)      // 108,800,000
#define AV_FLOATS  (8L*20L*240L*240L)       // 9,216,000
#define OUT_FP     80000L
#define OUT_MAP    9216000L
#define PI_F 3.14159265358979323846f

typedef __attribute__((ext_vector_type(16))) _Float16 v16h;
typedef __attribute__((ext_vector_type(8)))  float    v8f;
typedef __attribute__((ext_vector_type(8)))  float    v8fl;   // 32B load vector

// ---------------- zero workspace ----------------
__global__ void zero_kernel(float4* __restrict__ p, long n4) {
    long i = (long)blockIdx.x * blockDim.x + threadIdx.x;
    if (i < n4) p[i] = make_float4(0.f, 0.f, 0.f, 0.f);
}

// ---------------- trilinear scatter splat ----------------
__global__ void splat_kernel(const float* __restrict__ obs,
                             float* __restrict__ vox, float fcam) {
    long idx = (long)blockIdx.x * blockDim.x + threadIdx.x;
    if (idx >= (long)B_ * FH * FW) return;
    int col = (int)(idx % FW);
    int r   = (int)((idx / FW) % FH);
    int b   = (int)(idx / ((long)FW * FH));

    const long obsB = (long)b * C_ * FH * FW;
    const long pix  = (long)r * FW + col;
    float d = __builtin_nontemporal_load(obs + obsB + 3L * FH * FW + pix);

    float X = (col - (FW - 1) * 0.5f) * d / fcam + 250.0f;          // +VR*RES/2
    float Z = ((FH - 1 - r) - (FH - 1) * 0.5f) * d / fcam + 88.0f;  // +CAM_H_CM
    float Xn = (X / 5.0f - 50.0f) / 100.0f * 2.0f;
    float Yn = (d / 5.0f - 50.0f) / 100.0f * 2.0f;
    float Zn = (Z / 5.0f - 32.0f) / 80.0f * 2.0f;
    float pos[3] = { Xn * 50.0f + 50.0f, Yn * 50.0f + 50.0f, Zn * 40.0f + 40.0f };
    const float gdim[3] = { 100.0f, 100.0f, 80.0f };

    float pd[3][2], wd[3][2];
    #pragma unroll
    for (int dim = 0; dim < 3; ++dim) {
        float fl = floorf(pos[dim]);
        #pragma unroll
        for (int ix = 0; ix < 2; ++ix) {
            float p = fl + (float)ix;
            bool safe = (p > 0.0f) && (p < gdim[dim]);
            wd[dim][ix] = safe ? (1.0f - fabsf(pos[dim] - p)) : 0.0f;
            pd[dim][ix] = safe ? p : 0.0f;
        }
    }

    float feat[F_];
    feat[0] = 1.0f;
    #pragma unroll
    for (int f = 1; f < F_; ++f)
        feat[f] = __builtin_nontemporal_load(obs + obsB + (long)(3 + f) * FH * FW + pix);

    float* voxB = vox + (long)b * F_ * VOX_PER_CH;
    #pragma unroll
    for (int c0 = 0; c0 < 2; ++c0)
    #pragma unroll
    for (int c1 = 0; c1 < 2; ++c1)
    #pragma unroll
    for (int c2 = 0; c2 < 2; ++c2) {
        float w = wd[0][c0] * wd[1][c1] * wd[2][c2];
        if (w == 0.0f) continue;
        int vi = ((int)pd[0][c0] * VR_ + (int)pd[1][c1]) * ZH + (int)pd[2][c2];
        #pragma unroll
        for (int f = 0; f < F_; ++f)
            unsafeAtomicAdd(voxB + (long)f * VOX_PER_CH + vi, feat[f] * w);
    }
}

// ---------------- z-column reduction via WMMA ----------------
// One wave handles (b, f, x, ytile): 16 y-rows. A = rounded voxels (16x32 f16,
// M=y, K=z), B = ones => D[m,n] = sum_k A[m,k] in every column n.
// A layout (ISA 7.12.2): lane m (0-15) holds M=m; elements 0..7 -> K=e,
// 8..15 -> K=e+8; hi half-wave adds +8 to K. Per lane each chunk is two
// contiguous, 32B-aligned 8-float runs -> unconditional b128 loads, one wait.
__global__ __launch_bounds__(128) void reduce_wmma_kernel(
        const float* __restrict__ vox, float* __restrict__ av,
        float* __restrict__ outFp) {
    const int lane = threadIdx.x & 31;
    const int wid  = blockIdx.x * (blockDim.x >> 5) + (threadIdx.x >> 5);
    // wid = ((b*17 + f)*100 + x)*7 + ytile ; total 8*17*100*7 = 95200 waves exact
    const int ytile = wid % 7;
    int t = wid / 7;
    const int x = t % 100; t /= 100;
    const int f = t % 17;
    const int b = t / 17;
    const bool hi = lane >= 16;
    const int m  = lane & 15;
    const int y  = ytile * 16 + m;
    const int yc = (y < 100) ? y : 99;                 // clamp: always in-bounds
    const float rs = (y < 100) ? 1.0f : 0.0f;          // zero out pad rows
    const long base = (((long)(b * 17 + f) * 10000) + (long)x * 100) * 80;
    const float* rowp = vox + base + (long)yc * ZH + (hi ? 8 : 0);

    // 10x global_load_b128, no divergence (chunk2 runB is z>=80 -> const 0)
    v8fl r0a = __builtin_nontemporal_load((const v8fl*)(rowp +  0));
    v8fl r0b = __builtin_nontemporal_load((const v8fl*)(rowp + 16));
    v8fl r1a = __builtin_nontemporal_load((const v8fl*)(rowp + 32));
    v8fl r1b = __builtin_nontemporal_load((const v8fl*)(rowp + 48));
    v8fl r2a = __builtin_nontemporal_load((const v8fl*)(rowp + 64));

    v16h a0, a1, a2, ap;
    #pragma unroll
    for (int e = 0; e < 8; ++e) {
        a0[e]     = (_Float16)(rintf(r0a[e]) * rs);    // jnp.round == rne
        a0[e + 8] = (_Float16)(rintf(r0b[e]) * rs);
        a1[e]     = (_Float16)(rintf(r1a[e]) * rs);
        a1[e + 8] = (_Float16)(rintf(r1b[e]) * rs);
        a2[e]     = (_Float16)(rintf(r2a[e]) * rs);
        a2[e + 8] = (_Float16)0.0f;
    }
    // partial z-range [13,25) lives entirely in chunk 0:
    //  lo lanes: e=0..7 -> z=e (never);      e=8..15 -> z=e+8 in 16..23 (always)
    //  hi lanes: e=0..7 -> z=e+8 (e>=5);     e=8..15 -> z=e+16 (only e==8, z=24)
    #pragma unroll
    for (int e = 0; e < 16; ++e) {
        const bool inLo = (e >= 8);
        const bool inHi = (e < 8) ? (e >= 5) : (e == 8);
        const bool in   = hi ? inHi : inLo;
        ap[e] = in ? a0[e] : (_Float16)0.0f;
    }

    v16h ones;
    #pragma unroll
    for (int e = 0; e < 16; ++e) ones[e] = (_Float16)1.0f;

    v8f accAll  = {};
    v8f accPart = {};
    accPart = __builtin_amdgcn_wmma_f32_16x16x32_f16(
        false, ap, false, ones, (short)0, accPart, false, false);
    accAll = __builtin_amdgcn_wmma_f32_16x16x32_f16(
        false, a0, false, ones, (short)0, accAll, false, false);
    accAll = __builtin_amdgcn_wmma_f32_16x16x32_f16(
        false, a1, false, ones, (short)0, accAll, false, false);
    accAll = __builtin_amdgcn_wmma_f32_16x16x32_f16(
        false, a2, false, ones, (short)0, accAll, false, false);

    // lane 0 holds sums for rows 0..7, lane 16 for rows 8..15 (all cols equal)
    if ((lane & 15) == 0) {
        #pragma unroll
        for (int r = 0; r < 8; ++r) {
            int yy = ytile * 16 + r + (hi ? 8 : 0);
            if (yy >= 100) continue;
            float sPart = accPart[r];
            float sAll  = accAll[r];
            int ai = 120 + yy, aj = 70 + x;   // agent_view window
            if (f == 0) {
                float fp = fminf(fmaxf(sPart, 0.0f), 1.0f);       // /MAP_T=1
                outFp[((long)b * 100 + yy) * 100 + x] = fp;
                av[(((long)b * C_ + 0) * M_ + ai) * M_ + aj] = fp;
                av[(((long)b * C_ + 1) * M_ + ai) * M_ + aj] =
                    fminf(fmaxf(sAll, 0.0f), 1.0f);               // /EXP_T=1
            } else {
                av[(((long)b * C_ + 3 + f) * M_ + ai) * M_ + aj] =
                    fminf(fmaxf(sPart * 0.2f, 0.0f), 1.0f);       // /CAT_T=5
            }
        }
    }
}

// ---------------- pose update ----------------
__global__ void pose_kernel(const float* __restrict__ pose_obs,
                            const float* __restrict__ poses_last,
                            float* __restrict__ outP1, float* __restrict__ outP2,
                            float* __restrict__ pp) {
    int b = threadIdx.x;
    if (b >= B_) return;
    const float DEG = 57.29577951308232f;
    float ox = poses_last[b * 3 + 0], oy = poses_last[b * 3 + 1], ot = poses_last[b * 3 + 2];
    float dx = pose_obs[b * 3 + 0],  dy = pose_obs[b * 3 + 1],  dO = pose_obs[b * 3 + 2];
    float o  = ot / DEG;
    float ny = oy + dx * sinf(o) + dy * cosf(o);
    float nx = ox + dx * cosf(o) - dy * sinf(o);
    float nt = ot + dO * DEG;
    nt = fmodf(nt - 180.0f, 360.0f) + 180.0f;
    nt = fmodf(nt + 180.0f, 360.0f) - 180.0f;
    outP1[b * 3 + 0] = nx; outP1[b * 3 + 1] = ny; outP1[b * 3 + 2] = nt;
    outP2[b * 3 + 0] = nx; outP2[b * 3 + 1] = ny; outP2[b * 3 + 2] = nt;
    float stx = -(nx * 100.0f / 5.0f - 120.0f) / 120.0f;
    float sty = -(ny * 100.0f / 5.0f - 120.0f) / 120.0f;
    float stt = (90.0f - nt) * PI_F / 180.0f;
    pp[b * 4 + 0] = cosf(stt);
    pp[b * 4 + 1] = sinf(stt);
    pp[b * 4 + 2] = stx;
    pp[b * 4 + 3] = sty;
}

// ---------------- bilinear sampling (align_corners=True, zero pad) ----------------
__device__ inline float bilin240(const float* __restrict__ img, float sx, float sy) {
    float px = (sx + 1.0f) * 0.5f * (M_ - 1);
    float py = (sy + 1.0f) * 0.5f * (M_ - 1);
    float x0 = floorf(px), y0 = floorf(py);
    float wx = px - x0, wy = py - y0;
    float acc = 0.0f;
    #pragma unroll
    for (int ty = 0; ty < 2; ++ty)
    #pragma unroll
    for (int tx = 0; tx < 2; ++tx) {
        float xi = x0 + (float)tx, yi = y0 + (float)ty;
        float w = (tx ? wx : 1.0f - wx) * (ty ? wy : 1.0f - wy);
        if (xi >= 0.0f && xi < (float)M_ && yi >= 0.0f && yi < (float)M_)
            acc += img[(int)yi * M_ + (int)xi] * w;
    }
    return acc;
}

__global__ void warp_rot_kernel(const float* __restrict__ src,
                                const float* __restrict__ pp,
                                float* __restrict__ dst) {
    long idx = (long)blockIdx.x * blockDim.x + threadIdx.x;
    if (idx >= AV_FLOATS) return;
    int j = (int)(idx % M_);
    int i = (int)((idx / M_) % M_);
    long bc = idx / ((long)M_ * M_);          // b*20 + ch
    int b = (int)(bc / C_);
    float ct = pp[b * 4 + 0], st = pp[b * 4 + 1];
    float gx = ((float)j + 0.5f) * (2.0f / M_) - 1.0f;   // affine_grid, ac=False
    float gy = ((float)i + 0.5f) * (2.0f / M_) - 1.0f;
    float sx = ct * gx - st * gy;
    float sy = st * gx + ct * gy;
    dst[idx] = bilin240(src + bc * ((long)M_ * M_), sx, sy);
}

__global__ void warp_trans_kernel(const float* __restrict__ src,
                                  const float* __restrict__ pp,
                                  const float* __restrict__ maps_last,
                                  float* __restrict__ outMap) {
    long idx = (long)blockIdx.x * blockDim.x + threadIdx.x;
    if (idx >= AV_FLOATS) return;
    int j = (int)(idx % M_);
    int i = (int)((idx / M_) % M_);
    long bc = idx / ((long)M_ * M_);
    int b = (int)(bc / C_);
    float stx = pp[b * 4 + 2], sty = pp[b * 4 + 3];
    float gx = ((float)j + 0.5f) * (2.0f / M_) - 1.0f + stx;
    float gy = ((float)i + 0.5f) * (2.0f / M_) - 1.0f + sty;
    float v = bilin240(src + bc * ((long)M_ * M_), gx, gy);
    float ml = __builtin_nontemporal_load(maps_last + idx);
    outMap[idx] = fmaxf(ml, v);
}

// ---------------- launcher ----------------
extern "C" void kernel_launch(void* const* d_in, const int* in_sizes, int n_in,
                              void* d_out, int out_size, void* d_ws, size_t ws_size,
                              hipStream_t stream) {
    const float* obs        = (const float*)d_in[0];
    const float* pose_obs   = (const float*)d_in[1];
    const float* maps_last  = (const float*)d_in[2];
    const float* poses_last = (const float*)d_in[3];

    float* out     = (float*)d_out;
    float* outFp   = out;                       // [8,1,100,100]
    float* outMap  = out + OUT_FP;              // [8,20,240,240]
    float* outP1   = out + OUT_FP + OUT_MAP;    // [8,3]
    float* outP2   = outP1 + 24;                // [8,3]

    float* ws  = (float*)d_ws;
    float* vox = ws;                            // 108,800,000 f32
    float* av  = ws + VOX_FLOATS;               // 9,216,000 f32 (agent_view)
    float* rot = av + AV_FLOATS;                // 9,216,000 f32
    float* pp  = rot + AV_FLOATS;               // 32 f32 pose params

    float fcam = (float)(320.0 / tan(39.5 * 3.14159265358979323846 / 180.0));

    // 1) zero voxels + agent_view (float4 stores)
    long n4 = (VOX_FLOATS + AV_FLOATS) / 4;
    zero_kernel<<<(unsigned)((n4 + 255) / 256), 256, 0, stream>>>((float4*)vox, n4);

    // 2) trilinear scatter splat (atomic f32)
    long npix = (long)B_ * FH * FW;             // 2,457,600
    splat_kernel<<<(unsigned)((npix + 255) / 256), 256, 0, stream>>>(obs, vox, fcam);

    // 3) z reductions via v_wmma_f32_16x16x32_f16 (95,200 waves, 4 waves/block)
    reduce_wmma_kernel<<<23800, 128, 0, stream>>>(vox, av, outFp);

    // 4) pose update + warp params
    pose_kernel<<<1, 32, 0, stream>>>(pose_obs, poses_last, outP1, outP2, pp);

    // 5) rotation warp, 6) translation warp + max
    warp_rot_kernel<<<(unsigned)((AV_FLOATS + 255) / 256), 256, 0, stream>>>(av, pp, rot);
    warp_trans_kernel<<<(unsigned)((AV_FLOATS + 255) / 256), 256, 0, stream>>>(rot, pp, maps_last, outMap);
}
